// Mamba2Layer_90331752170261
// MI455X (gfx1250) — matrix operand order repaired and checked by reading, compiled, not run on hardware
//
#include <hip/hip_runtime.h>
#include <stdint.h>

// ---------------------------------------------------------------------------
// Mamba2-style RNN layer for gfx1250 (MI455X), wave32 + WMMA f16 + async LDS.
//   xproj = x @ Wx + b1            (WMMA GEMM, f16 in / f32 acc)
//   h_t   = gelu(xproj_t + h@Wh)   (persistent scan, Wh register-resident,
//                                   xproj double-buffered via async-to-LDS)
//   y     = hs @ W2 + b2           (WMMA GEMM, A staged via async-to-LDS)
// ---------------------------------------------------------------------------

typedef __attribute__((ext_vector_type(16))) _Float16 v16h;
typedef __attribute__((ext_vector_type(8)))  _Float16 v8h;
typedef __attribute__((ext_vector_type(8)))  float    v8f;
typedef __attribute__((ext_vector_type(4)))  float    v4f;

#define INPUT_DIM 256
#define HIDDEN    512
#define OUT_DIM   256
#define BATCH     32
#define SEQ       4096

static constexpr size_t XPROJ_BYTES = (size_t)SEQ * BATCH * HIDDEN * 4;
static constexpr size_t HS_BYTES    = (size_t)SEQ * BATCH * HIDDEN * 2;

__device__ __forceinline__ float gelu_tanh(float x) {
    // matches jax.nn.gelu (approximate=True)
    float x3 = x * x * x;
    return 0.5f * x * (1.0f + tanhf(0.7978845608028654f * (x + 0.044715f * x3)));
}

__device__ __forceinline__ v8f wmma16(v16h a, v16h b, v8f c) {
    return __builtin_amdgcn_wmma_f32_16x16x32_f16(
        false, a, false, b, (short)0, c, false, false);
}

// --- gfx1250 async LDS<->global copies (ASYNCcnt-tracked) -------------------
__device__ __forceinline__ uint32_t lds_off32(const void* p) {
    // generic shared pointer = {SHARED_BASE[63:32], lds_byte_offset[31:0]}
    return (uint32_t)(uintptr_t)p;
}
__device__ __forceinline__ void async_load_b128(uint32_t lds, const void* gaddr) {
    asm volatile("global_load_async_to_lds_b128 %0, %1, off"
                 :: "v"(lds), "v"(gaddr) : "memory");
}
__device__ __forceinline__ void async_store_b128(const void* gaddr, uint32_t lds) {
    asm volatile("global_store_async_from_lds_b128 %0, %1, off"
                 :: "v"(gaddr), "v"(lds) : "memory");
}
__device__ __forceinline__ void wait_async0() {
    asm volatile("s_wait_asynccnt 0x0" ::: "memory");
}

// A-fragment (16x32 f16, MxK) from LDS tile, row stride `stride` halves.
//   lanes 0-15:  M=lane,    K = kc*32 + {0..7, 16..23}
//   lanes 16-31: M=lane-16, K = kc*32 + {8..15, 24..31}
__device__ __forceinline__ v16h lds_afrag(const _Float16* tile, int m, int khalf,
                                          int kc, int stride) {
    const _Float16* p = tile + m * stride + kc * 32 + khalf * 8;
    v8h lo = *(const v8h*)p;
    v8h hi = *(const v8h*)(p + 16);
    v16h a;
#pragma unroll
    for (int j = 0; j < 8; j++) { a[j] = lo[j]; a[8 + j] = hi[j]; }
    return a;
}

// B-fragment (32x16 f16, KxN) from pre-transposed [n][k] weights, stride K.
//   lanes 0-15:  N=lane,    K = kc*32 + {0..7, 16..23}
//   lanes 16-31: N=lane-16, K = kc*32 + {8..15, 24..31}
__device__ __forceinline__ v16h gbl_bfrag(const _Float16* wT, int n, int khalf,
                                          int kc, int K) {
    const _Float16* p = wT + (size_t)n * K + kc * 32 + khalf * 8;
    v8h lo = *(const v8h*)p;
    v8h hi = *(const v8h*)(p + 16);
    v16h b;
#pragma unroll
    for (int j = 0; j < 8; j++) { b[j] = lo[j]; b[8 + j] = hi[j]; }
    return b;
}

// ---------------------------------------------------------------------------
// Kernel 0: transpose + f32->f16 weight conversion (one-shot, tiny).
// ---------------------------------------------------------------------------
__global__ __launch_bounds__(256) void wconv_kernel(
    const float* __restrict__ W1, const float* __restrict__ W2,
    _Float16* __restrict__ WxT, _Float16* __restrict__ WhT,
    _Float16* __restrict__ W2T) {
    int id = blockIdx.x * 256 + threadIdx.x;
    if (id < 512 * 256) {
        int n = id >> 8, k = id & 255;
        WxT[id] = (_Float16)W1[k * HIDDEN + n];
    } else if (id < 512 * 256 + 512 * 512) {
        int j = id - 512 * 256;
        int n = j >> 9, k = j & 511;
        WhT[j] = (_Float16)W1[(INPUT_DIM + k) * HIDDEN + n];
    } else if (id < 512 * 256 + 512 * 512 + 256 * 512) {
        int j = id - 512 * 256 - 512 * 512;
        int o = j >> 9, h = j & 511;
        W2T[j] = (_Float16)W2[h * OUT_DIM + o];
    }
}

// ---------------------------------------------------------------------------
// Kernel 1: xproj[t*32+b][h] = x[b][t][:] @ Wx + b1.
// Block = 16 rows x 512 cols, 8 waves, wave owns 64 cols.  (x needs f32->f16
// conversion so A staging stays on the VALU path.)
// ---------------------------------------------------------------------------
__global__ __launch_bounds__(256) void xproj_kernel(
    const float* __restrict__ x, const float* __restrict__ b1,
    const _Float16* __restrict__ WxT, float* __restrict__ xproj) {
    __shared__ __align__(16) _Float16 aTile[16][264];   // 16 x 256, padded

    const int r0  = blockIdx.x * 16;        // global row = t*32 + b
    const int tid = threadIdx.x;
    {
        int row = tid >> 4, seg = tid & 15;
        int r = r0 + row;
        int b = r & 31, t = r >> 5;
        const float* xr = x + ((size_t)b * SEQ + t) * INPUT_DIM + seg * 16;
#pragma unroll
        for (int j = 0; j < 16; j += 4) {
            v4f v = *(const v4f*)(xr + j);
#pragma unroll
            for (int e = 0; e < 4; e++)
                aTile[row][seg * 16 + j + e] = (_Float16)v[e];
        }
    }
    __syncthreads();

    const int lane = tid & 31, wave = tid >> 5;
    const int nbase = wave * 64;
    const int m = lane & 15, khalf = lane >> 4;

    v8f acc[4];
#pragma unroll
    for (int nt = 0; nt < 4; nt++) {
        float bv = b1[nbase + nt * 16 + m];
#pragma unroll
        for (int v = 0; v < 8; v++) acc[nt][v] = bv;
    }

#pragma unroll
    for (int kc = 0; kc < 8; kc++) {
        v16h a = lds_afrag(&aTile[0][0], m, khalf, kc, 264);
#pragma unroll
        for (int nt = 0; nt < 4; nt++) {
            v16h bf = gbl_bfrag(WxT, nbase + nt * 16 + m, khalf, kc, INPUT_DIM);
            acc[nt] = wmma16(a, bf, acc[nt]);
        }
    }

#pragma unroll
    for (int nt = 0; nt < 4; nt++) {
        int n = nbase + nt * 16 + m;
#pragma unroll
        for (int v = 0; v < 8; v++)
            xproj[(size_t)(r0 + v + 8 * khalf) * HIDDEN + n] = acc[nt][v];
    }
}

// ---------------------------------------------------------------------------
// Kernel 2: persistent recurrence.  2 WGs x 512 threads (16 waves).
// Wave owns a 32-wide N-slice of Wh, register-resident (32 x v16h).
// h ping-pongs in LDS; xproj[t+1] is async-prefetched into an LDS ping-pong
// during step t's WMMAs; h_{t-1} drains to hs via async stores.
// One s_wait_asynccnt + one barrier per step.
// ---------------------------------------------------------------------------
#define HSTR 520   // h row stride (halves):  1040 B, conflict-free b128 reads
#define XSTR 516   // xproj row stride (f32): 2064 B, half-waves on disjoint banks

__global__ __launch_bounds__(512) void scan_kernel(
    const float* __restrict__ xproj, const _Float16* __restrict__ WhT,
    _Float16* __restrict__ hs) {
    __shared__ __align__(16) _Float16 hbuf[2][16 * HSTR];
    __shared__ __align__(16) float    xbuf[2][16 * XSTR];

    const int tid  = threadIdx.x;
    const int lane = tid & 31, wave = tid >> 5;   // 16 waves
    const int b0   = blockIdx.x * 16;             // batch rows of this WG
    const int nw   = wave * 32;
    const int m = lane & 15, khalf = lane >> 4;
    const int crow = tid >> 5;                    // copy row (== wave)

    // Register-resident Wh fragments: 16 kc x 2 n-tiles = 256 VGPRs.
    v16h bwh[16][2];
#pragma unroll
    for (int kc = 0; kc < 16; kc++)
#pragma unroll
        for (int nt = 0; nt < 2; nt++)
            bwh[kc][nt] = gbl_bfrag(WhT, nw + nt * 16 + m, khalf, kc, HIDDEN);

    // h_{-1} = 0, and async-stage xproj[0].
    for (int i = tid; i < 16 * HSTR; i += 512) hbuf[0][i] = (_Float16)0.0f;
    {
        const char* g = (const char*)(xproj + ((size_t)0 * BATCH + b0 + crow) * HIDDEN)
                        + lane * 64;
        uint32_t l = lds_off32(&xbuf[0][crow * XSTR]) + lane * 64;
#pragma unroll
        for (int j = 0; j < 4; j++) async_load_b128(l + 16 * j, g + 16 * j);
    }
    wait_async0();
    __syncthreads();

    for (int t = 0; t < SEQ; t++) {
        const _Float16* cur = hbuf[t & 1];
        _Float16*       nxt = hbuf[(t + 1) & 1];
        const float*    xb  = xbuf[t & 1];

        // Async-prefetch xproj[t+1] into the other x-buffer (32 KB, contiguous).
        if (t + 1 < SEQ) {
            const char* g =
                (const char*)(xproj + ((size_t)(t + 1) * BATCH + b0 + crow) * HIDDEN)
                + lane * 64;
            uint32_t l = lds_off32(&xbuf[(t + 1) & 1][crow * XSTR]) + lane * 64;
#pragma unroll
            for (int j = 0; j < 4; j++) async_load_b128(l + 16 * j, g + 16 * j);
        }
        if (t + 2 < SEQ)   // warm L2 one more step ahead
            __builtin_prefetch(xproj + ((size_t)(t + 2) * BATCH + b0 + m) * HIDDEN + nw,
                               0, 1);

        // Drain h_{t-1} -> hs[t-1] straight from LDS (async, overlaps WMMAs).
        if (t > 0) {
            const char* g =
                (const char*)(hs + ((size_t)(t - 1) * BATCH + b0 + crow) * HIDDEN)
                + lane * 32;
            uint32_t l = lds_off32(cur + crow * HSTR) + lane * 32;
            async_store_b128(g, l);
            async_store_b128(g + 16, l + 16);
        }

        // acc = xproj[t] tile from LDS (C-fragment layout).
        v8f acc[2];
#pragma unroll
        for (int nt = 0; nt < 2; nt++) {
            int n = nw + nt * 16 + m;
#pragma unroll
            for (int v = 0; v < 8; v++)
                acc[nt][v] = xb[(v + 8 * khalf) * XSTR + n];
        }

        // acc += h_{t-1} @ Wh   (A from LDS, B from registers)
#pragma unroll
        for (int kc = 0; kc < 16; kc++) {
            v16h a = lds_afrag(cur, m, khalf, kc, HSTR);
            acc[0] = wmma16(a, bwh[kc][0], acc[0]);
            acc[1] = wmma16(a, bwh[kc][1], acc[1]);
        }

        // gelu, write h_t into the other buffer.
#pragma unroll
        for (int nt = 0; nt < 2; nt++) {
            int n = nw + nt * 16 + m;
#pragma unroll
            for (int v = 0; v < 8; v++)
                nxt[(v + 8 * khalf) * HSTR + n] = (_Float16)gelu_tanh(acc[nt][v]);
        }

        wait_async0();     // xbuf[t+1] landed, hs drain done, before reuse
        __syncthreads();
    }

    // Final state h_{SEQ-1} lives in hbuf[SEQ & 1] == hbuf[0].
    {
        const _Float16* s = hbuf[0] + crow * HSTR + lane * 16;
        _Float16* d = hs + ((size_t)(SEQ - 1) * BATCH + b0 + crow) * HIDDEN + lane * 16;
        *(v8h*)d       = *(const v8h*)s;
        *(v8h*)(d + 8) = *(const v8h*)(s + 8);
    }
}

// ---------------------------------------------------------------------------
// Kernel 3: y[b][t][o] = hs[t*32+b][:] @ W2 + b2.
// Block = 16 rows x 256 cols, 8 waves; A tile staged via async-to-LDS.
// ---------------------------------------------------------------------------
__global__ __launch_bounds__(256) void ygemm_kernel(
    const _Float16* __restrict__ hs, const _Float16* __restrict__ W2T,
    const float* __restrict__ b2, float* __restrict__ y) {
    __shared__ __align__(16) _Float16 aTile[16][HSTR];   // 16 x 512, padded

    const int r0  = blockIdx.x * 16;
    const int tid = threadIdx.x;
    {
        int row = tid >> 4, seg = tid & 15;              // 64 B per thread
        const char* g = (const char*)(hs + (size_t)(r0 + row) * HIDDEN) + seg * 64;
        uint32_t l = lds_off32(&aTile[row][seg * 32]);
#pragma unroll
        for (int j = 0; j < 4; j++) async_load_b128(l + 16 * j, g + 16 * j);
    }
    wait_async0();
    __syncthreads();

    const int lane = tid & 31, wave = tid >> 5;
    const int nbase = wave * 32;
    const int m = lane & 15, khalf = lane >> 4;

    v8f acc[2];
#pragma unroll
    for (int nt = 0; nt < 2; nt++) {
        float bv = b2[nbase + nt * 16 + m];
#pragma unroll
        for (int v = 0; v < 8; v++) acc[nt][v] = bv;
    }

#pragma unroll
    for (int kc = 0; kc < 16; kc++) {
        v16h a = lds_afrag(&aTile[0][0], m, khalf, kc, HSTR);
#pragma unroll
        for (int nt = 0; nt < 2; nt++) {
            v16h bf = gbl_bfrag(W2T, nbase + nt * 16 + m, khalf, kc, HIDDEN);
            acc[nt] = wmma16(a, bf, acc[nt]);
        }
    }

#pragma unroll
    for (int nt = 0; nt < 2; nt++) {
        int n = nbase + nt * 16 + m;
#pragma unroll
        for (int v = 0; v < 8; v++) {
            int r = r0 + v + 8 * khalf;
            int b = r & 31, t = r >> 5;
            y[((size_t)b * SEQ + t) * OUT_DIM + n] = acc[nt][v];
        }
    }
}

// ---------------------------------------------------------------------------
extern "C" void kernel_launch(void* const* d_in, const int* in_sizes, int n_in,
                              void* d_out, int out_size, void* d_ws, size_t ws_size,
                              hipStream_t stream) {
    (void)in_sizes; (void)n_in; (void)out_size; (void)ws_size;
    const float* x  = (const float*)d_in[0];
    const float* W1 = (const float*)d_in[1];
    const float* b1 = (const float*)d_in[2];
    const float* W2 = (const float*)d_in[3];
    const float* b2 = (const float*)d_in[4];
    float* y = (float*)d_out;

    char* ws = (char*)d_ws;
    float*    xproj = (float*)ws;                                  // 256 MB
    _Float16* hs    = (_Float16*)(ws + XPROJ_BYTES);               // 128 MB
    _Float16* WxT   = (_Float16*)(ws + XPROJ_BYTES + HS_BYTES);    // 256 KB
    _Float16* WhT   = WxT + 512 * 256;                             // 512 KB
    _Float16* W2T   = WhT + 512 * 512;                             // 256 KB

    wconv_kernel<<<2048, 256, 0, stream>>>(W1, W2, WxT, WhT, W2T);
    xproj_kernel<<<(SEQ * BATCH) / 16, 256, 0, stream>>>(x, b1, WxT, xproj);
    scan_kernel<<<2, 512, 0, stream>>>(xproj, WhT, hs);
    ygemm_kernel<<<(SEQ * BATCH) / 16, 256, 0, stream>>>(hs, W2T, b2, y);
}
